// MultiHeadAttention_26792005993026
// MI455X (gfx1250) — matrix operand order repaired and checked by reading, not compiled, not run on hardware
//
#include <hip/hip_runtime.h>
#include <hip/hip_bf16.h>

// Problem constants (from reference)
#define B_  32
#define T_  4096
#define D_  512      // D_IN (K dimension)
#define H_  512      // HID  (N dimension)
#define TM  64       // rows per workgroup tile
#define KB  256      // K columns staged in LDS per phase
#define MTILES   (TM / 16)        // 4
#define NT_WAVE  4                // 16-col tiles per wave (8 waves * 64 = 512 cols)
#define CHUNKS   (T_ / TM)        // 64 chunks per batch
#define LDSK     (KB + 8)         // padded LDS row stride (halves)

typedef __attribute__((ext_vector_type(16))) _Float16 v16h;
typedef __attribute__((ext_vector_type(8)))  _Float16 v8h;
typedef __attribute__((ext_vector_type(8)))  float    v8f;
typedef __attribute__((ext_vector_type(4)))  float    v4f;

// ---- fragment loaders ----------------------------------------------------
// A-matrix 16x32 f16 fragment as loaded here (contiguous k, matching B): lanes 0-15 hold row M=lane with
// K = {k0..k0+15}; lanes 16-31 hold row M=lane-16 with
// K = {k0+16..k0+31}.  Two 16B ds loads per fragment.
__device__ __forceinline__ v16h load_a_frag(const _Float16* sx, int m0, int k0,
                                            int lan16, int hi) {
    const _Float16* rowp = sx + (m0 + lan16) * LDSK + k0 + (hi ? 16 : 0);
    v8h lo  = *(const v8h*)(rowp);
    v8h hi8 = *(const v8h*)(rowp + 8);
    v16h a;
#pragma unroll
    for (int e = 0; e < 8; ++e) { a[e] = lo[e]; a[e + 8] = hi8[e]; }
    return a;
}

// B-matrix 32x16 f16 layout: lanes 0-15 hold col N=lane with K=k..k+15,
// lanes 16-31 hold col N=lane-16 with K=k+16..k+31. W pre-converted to f16
// (row-major [H][D]) so this is a single contiguous 32B load per lane from L2.
__device__ __forceinline__ v16h load_b_frag(const _Float16* __restrict__ Wh,
                                            int n0, int k, int lan16, int hi) {
    return *(const v16h*)(Wh + (size_t)(n0 + lan16) * D_ + k + (hi ? 16 : 0));
}

// Cooperative stage of a TM x KB f32 sub-tile of x into LDS as f16.
__device__ __forceinline__ void stage_x(const float* __restrict__ x,
                                        size_t xbase, int kbase,
                                        _Float16* sx, int tid) {
    const int NV4 = TM * KB / 4;
#pragma unroll 4
    for (int i = tid; i < NV4; i += 256) {
        int row = i / (KB / 4);
        int c4  = i % (KB / 4);
        v4f f = *(const v4f*)(x + xbase + (size_t)row * D_ + kbase + c4 * 4);
        _Float16* dst = sx + row * LDSK + c4 * 4;
        dst[0] = (_Float16)f.x; dst[1] = (_Float16)f.y;
        dst[2] = (_Float16)f.z; dst[3] = (_Float16)f.w;
    }
}

// Core GEMM: acc[mi][ni] += A(64x512 tile) x B(512x64 stripe), K-phased.
__device__ __forceinline__ void gemm_tile(const float* __restrict__ x,
                                          size_t xbase,
                                          const _Float16* __restrict__ Wh,
                                          _Float16* sx,
                                          int tid, int wave, int lan16, int hi,
                                          v8f acc[MTILES][NT_WAVE]) {
    for (int ph = 0; ph < D_ / KB; ++ph) {
        __syncthreads();                       // previous-phase reads done
        stage_x(x, xbase, ph * KB, sx, tid);
        __syncthreads();
#pragma unroll 2
        for (int k0 = 0; k0 < KB; k0 += 32) {
            v16h a[MTILES];
#pragma unroll
            for (int mi = 0; mi < MTILES; ++mi)
                a[mi] = load_a_frag(sx, mi * 16, k0, lan16, hi);
#pragma unroll
            for (int ni = 0; ni < NT_WAVE; ++ni) {
                v16h b = load_b_frag(Wh, wave * 64 + ni * 16,
                                     ph * KB + k0, lan16, hi);
#pragma unroll
                for (int mi = 0; mi < MTILES; ++mi)
                    acc[mi][ni] = __builtin_amdgcn_wmma_f32_16x16x32_f16(
                        false, a[mi], false, b, (short)0, acc[mi][ni],
                        false, false);
            }
        }
    }
}

// ---- Pass 0: one-shot W f32 -> f16 conversion (W reused by 4096 blocks) --
__global__ __launch_bounds__(256)
void k_prep(const float* __restrict__ W, _Float16* __restrict__ Wh) {
    int gid = blockIdx.x * 256 + threadIdx.x;     // H_*D_/4 threads
    v4f f = *(const v4f*)(W + (size_t)gid * 4);
    _Float16* d = Wh + (size_t)gid * 4;
    d[0] = (_Float16)f.x; d[1] = (_Float16)f.y;
    d[2] = (_Float16)f.z; d[3] = (_Float16)f.w;
}

// ---- Pass 1: scores[b,t] = V . tanh(x W^T + Wb) + Vb ---------------------
__global__ __launch_bounds__(256)
void k_scores(const float* __restrict__ x, const _Float16* __restrict__ Wh,
              const float* __restrict__ Wb, const float* __restrict__ Vw,
              const float* __restrict__ Vb, float* __restrict__ scores) {
    __shared__ _Float16 sx[TM * LDSK];
    __shared__ float ssc[TM];

    const int tid   = threadIdx.x;
    const int chunk = blockIdx.x % CHUNKS;
    const int b     = blockIdx.x / CHUNKS;
    const int t0    = chunk * TM;
    if (tid < TM) ssc[tid] = 0.f;

    const int wave = tid >> 5, lane = tid & 31;
    const int lan16 = lane & 15, hi = lane >> 4;

    v8f acc[MTILES][NT_WAVE] = {};
    gemm_tile(x, ((size_t)b * T_ + t0) * D_, Wh, sx, tid, wave, lan16, hi, acc);

    // Epilogue: tanh + dot with V per row; C/D layout: lane holds col
    // n = wave*64+ni*16+lan16, rows mi*16 + hi*8 + r (r = acc VGPR index).
    float prow[MTILES][8];
#pragma unroll
    for (int mi = 0; mi < MTILES; ++mi)
#pragma unroll
        for (int r = 0; r < 8; ++r) prow[mi][r] = 0.f;

#pragma unroll
    for (int ni = 0; ni < NT_WAVE; ++ni) {
        int n = wave * 64 + ni * 16 + lan16;
        float bias = Wb[n], vw = Vw[n];
#pragma unroll
        for (int mi = 0; mi < MTILES; ++mi)
#pragma unroll
            for (int r = 0; r < 8; ++r) {
                float e = tanhf(acc[mi][ni][r] + bias);
                prow[mi][r] += e * vw;
            }
    }
    // Reduce over the 16 lanes of each half-wave group (same rows, diff cols)
#pragma unroll
    for (int mi = 0; mi < MTILES; ++mi)
#pragma unroll
        for (int r = 0; r < 8; ++r) {
            float p = prow[mi][r];
            p += __shfl_xor(p, 1, 32);
            p += __shfl_xor(p, 2, 32);
            p += __shfl_xor(p, 4, 32);
            p += __shfl_xor(p, 8, 32);
            if (lan16 == 0)
                atomicAdd(&ssc[mi * 16 + hi * 8 + r], p);  // ds_add_f32
        }
    __syncthreads();
    if (tid < TM)
        scores[(size_t)b * T_ + t0 + tid] = ssc[tid] + Vb[0];
}

// ---- Pass 2: softmax over T per batch (in-place on workspace) ------------
__global__ __launch_bounds__(256)
void k_softmax(float* __restrict__ sc) {
    __shared__ float red[256];
    const int b = blockIdx.x, tid = threadIdx.x;
    float v[T_ / 256];
    float m = -3.0e38f;
#pragma unroll
    for (int i = 0; i < T_ / 256; ++i) {
        v[i] = sc[(size_t)b * T_ + tid + i * 256];
        m = fmaxf(m, v[i]);
    }
    red[tid] = m; __syncthreads();
    for (int s = 128; s > 0; s >>= 1) {
        if (tid < s) red[tid] = fmaxf(red[tid], red[tid + s]);
        __syncthreads();
    }
    m = red[0]; __syncthreads();
    float sum = 0.f;
#pragma unroll
    for (int i = 0; i < T_ / 256; ++i) { v[i] = __expf(v[i] - m); sum += v[i]; }
    red[tid] = sum; __syncthreads();
    for (int s = 128; s > 0; s >>= 1) {
        if (tid < s) red[tid] += red[tid + s];
        __syncthreads();
    }
    float inv = 1.f / red[0];
#pragma unroll
    for (int i = 0; i < T_ / 256; ++i)
        sc[(size_t)b * T_ + tid + i * 256] = v[i] * inv;
}

// ---- Pass 3: recompute energy tile, weight by attn, partial column sums --
__global__ __launch_bounds__(256)
void k_ctx(const float* __restrict__ x, const _Float16* __restrict__ Wh,
           const float* __restrict__ Wb, const float* __restrict__ attn,
           float* __restrict__ partials) {
    __shared__ _Float16 sx[TM * LDSK];
    __shared__ float s_attn[TM];

    const int tid   = threadIdx.x;
    const int chunk = blockIdx.x % CHUNKS;
    const int b     = blockIdx.x / CHUNKS;
    const int t0    = chunk * TM;
    if (tid < TM) s_attn[tid] = attn[(size_t)b * T_ + t0 + tid];

    const int wave = tid >> 5, lane = tid & 31;
    const int lan16 = lane & 15, hi = lane >> 4;

    v8f acc[MTILES][NT_WAVE] = {};
    gemm_tile(x, ((size_t)b * T_ + t0) * D_, Wh, sx, tid, wave, lan16, hi, acc);

    float ctxp[NT_WAVE] = {0.f, 0.f, 0.f, 0.f};
#pragma unroll
    for (int ni = 0; ni < NT_WAVE; ++ni) {
        int n = wave * 64 + ni * 16 + lan16;
        float bias = Wb[n];
#pragma unroll
        for (int mi = 0; mi < MTILES; ++mi)
#pragma unroll
            for (int r = 0; r < 8; ++r) {
                float e = tanhf(acc[mi][ni][r] + bias);
                ctxp[ni] += s_attn[mi * 16 + hi * 8 + r] * e;
            }
    }
    // lanes L and L+16 hold the same column (different row halves): combine.
#pragma unroll
    for (int ni = 0; ni < NT_WAVE; ++ni) {
        float v = ctxp[ni];
        v += __shfl_xor(v, 16, 32);
        if (!hi)
            partials[((size_t)b * CHUNKS + chunk) * H_ +
                     wave * 64 + ni * 16 + lan16] = v;
    }
}

// ---- Pass 4: reduce partials over chunks -> ctx[b, h] --------------------
__global__ __launch_bounds__(256)
void k_reduce(const float* __restrict__ partials, float* __restrict__ out) {
    int gid = blockIdx.x * 256 + threadIdx.x;
    if (gid >= B_ * H_) return;
    int b = gid / H_, h = gid % H_;
    float s = 0.f;
#pragma unroll 8
    for (int c = 0; c < CHUNKS; ++c)
        s += partials[((size_t)b * CHUNKS + c) * H_ + h];
    out[gid] = s;
}

extern "C" void kernel_launch(void* const* d_in, const int* in_sizes, int n_in,
                              void* d_out, int out_size, void* d_ws, size_t ws_size,
                              hipStream_t stream) {
    const float* x  = (const float*)d_in[0];
    const float* Ww = (const float*)d_in[1];
    const float* Wb = (const float*)d_in[2];
    const float* Vw = (const float*)d_in[3];
    const float* Vb = (const float*)d_in[4];
    float* out = (float*)d_out;

    // workspace: [Wh f16: 512KB][scores/attn f32: 512KB][partials f32: 4MB]
    _Float16* Wh     = (_Float16*)d_ws;
    float*    scores = (float*)((char*)d_ws + (size_t)H_ * D_ * sizeof(_Float16));
    float*    partials = scores + (size_t)B_ * T_;

    dim3 blk(256);
    k_prep   <<<(H_ * D_ / 4) / 256, blk, 0, stream>>>(Ww, Wh);
    k_scores <<<B_ * CHUNKS, blk, 0, stream>>>(x, Wh, Wb, Vw, Vb, scores);
    k_softmax<<<B_,          blk, 0, stream>>>(scores);
    k_ctx    <<<B_ * CHUNKS, blk, 0, stream>>>(x, Wh, Wb, scores, partials);
    k_reduce <<<(B_ * H_ + 255) / 256, blk, 0, stream>>>(partials, out);
}
